// DirectTransitionNet_43662637531853
// MI455X (gfx1250) — matrix operand order repaired and checked by reading, not compiled, not run on hardware
//
#include <hip/hip_runtime.h>
#include <cstdint>
#include <cstddef>

#define BATCH 32768
#define LDIM  16
#define CDIM  8
#define FIN   25
#define DDIM  64
#define NHEAD 4
#define HDIM  16
#define HID   512
#define FEAT  (FIN * DDIM) /* 1600 */

typedef __attribute__((ext_vector_type(16))) __bf16 v16bf;
typedef __attribute__((ext_vector_type(8)))  float  v8f;

union BF16Frag { v16bf v; unsigned short s[16]; uint4 q[2]; };

__device__ __forceinline__ unsigned short f2bf(float f) {
  union { float f; unsigned u; } cv; cv.f = f;
  unsigned r = cv.u + 0x7FFFu + ((cv.u >> 16) & 1u);
  return (unsigned short)(r >> 16);
}

// ---- WMMA fragment-layout index helpers -----------------------------------
// A-subtile is 16(m) x 32(k).  lane = m | ((k>>4)&1)<<4 ; elem = k & 15
__device__ __forceinline__ int a_lane(int m, int k) { return (m & 15) | (((k >> 4) & 1) << 4); }
__device__ __forceinline__ int a_elem(int k)        { return k & 15; }
// B-subtile is 32(k) x 16(n).  lane = n | ((k>>4)&1)<<4 ; elem = k & 15
__device__ __forceinline__ int b_lane(int n, int k) { return (n & 15) | (((k >> 4) & 1) << 4); }
__device__ __forceinline__ int b_elem(int k)        { return k & 15; }

// Fragment-tiled GLOBAL layout for activations (A operand), tile = 128m x 32k.
// Tile is 8 contiguous subtiles of [lane][16]; tiles ordered (mblk, kblk) row-major.
__device__ __forceinline__ size_t afrag_idx(size_t m, int k, int K) {
  const size_t tile = (m >> 7) * (size_t)(K >> 5) + (size_t)(k >> 5);
  const int msub = (int)((m >> 4) & 7);
  const int kk = k & 31;
  return (tile * 8 + msub) * 512 + (size_t)a_lane((int)(m & 15), kk) * 16 + a_elem(kk);
}
// Fragment-tiled GLOBAL layout for weights (B operand), tile = 32k x 64n.
__device__ __forceinline__ size_t bfrag_idx(int n, int k, int K) {
  const size_t tile = (size_t)(n >> 6) * (size_t)(K >> 5) + (size_t)(k >> 5);
  const int nsub = (n >> 4) & 3;
  const int kk = k & 31;
  return (tile * 4 + nsub) * 512 + (size_t)b_lane(n & 15, kk) * 16 + b_elem(kk);
}

// Load one fragment (16 bf16 per lane, contiguous 32B) from fragment-layout LDS.
__device__ __forceinline__ v16bf load_frag(const unsigned short* buf, int sub, int lane) {
  BF16Frag f;
  const uint4* p = reinterpret_cast<const uint4*>(buf + sub * 512 + lane * 16);
  f.q[0] = p[0];
  f.q[1] = p[1];
  return f.v;
}

// B-fragment directly from N-major (transposed) global weights: contiguous 32B/lane.
__device__ __forceinline__ v16bf load_frag_bt(const unsigned short* Bt, int ldk,
                                              int k0, int n0, int lane) {
  const int n = lane & 15, hi = lane >> 4;
  BF16Frag f;
  const uint4* p = reinterpret_cast<const uint4*>(Bt + (size_t)(n0 + n) * ldk + k0 + hi * 16);
  f.q[0] = p[0];
  f.q[1] = p[1];
  return f.v;
}

__device__ __forceinline__ v8f wmma_bf16(v16bf a, v16bf b, v8f c) {
  return __builtin_amdgcn_wmma_f32_16x16x32_bf16(false, a, false, b, (short)0, c,
                                                 false, false);
}

__device__ __forceinline__ v8f v8f_zero() {
  v8f z = {0.f, 0.f, 0.f, 0.f, 0.f, 0.f, 0.f, 0.f};
  return z;
}

__device__ __forceinline__ float softplusf(float x) {
  return (x > 20.f) ? x : log1pf(__expf(x));
}

// ---- gfx1250 async memory->LDS copy (ASYNCcnt tracked, no VGPR data) ------
// VDST = LDS byte offset (low 32 bits of generic LDS pointer), VADDR = 64b addr.
__device__ __forceinline__ void async_copy_b128(const void* gsrc, void* lds) {
  const unsigned ldsoff = (unsigned)(size_t)lds;
  asm volatile("global_load_async_to_lds_b128 %0, %1, off"
               :: "v"(ldsoff), "v"(gsrc)
               : "memory");
}
__device__ __forceinline__ void wait_async0() {
  asm volatile("s_wait_asynccnt 0x0" ::: "memory");
}

// ---------------------------------------------------------------------------
// Weight conversion: f32 (KxN row-major) -> bf16 transposed (NxK), for the
// small front-end weights consumed straight from global.
// ---------------------------------------------------------------------------
__global__ __launch_bounds__(256) void cvt_bf16_t_kernel(const float* __restrict__ in,
                                                         unsigned short* __restrict__ out,
                                                         int K, int N) {
  const int i = blockIdx.x * 256 + threadIdx.x;
  if (i < K * N) {
    const int k = i / N, n = i % N;
    out[(size_t)n * K + k] = f2bf(in[i]);
  }
}

// Weight conversion: f32 (KxN row-major) -> bf16 fragment-tiled (for async GEMM)
__global__ __launch_bounds__(256) void cvt_bf16_frag_kernel(const float* __restrict__ in,
                                                            unsigned short* __restrict__ out,
                                                            int K, int N) {
  const int i = blockIdx.x * 256 + threadIdx.x;
  if (i < K * N) {
    const int k = i / N, n = i % N;
    out[bfrag_idx(n, k, K)] = f2bf(in[i]);
  }
}

// ---------------------------------------------------------------------------
// Fused front-end: embed -> qkv (WMMA) -> attention -> w_o (WMMA) -> LN
// One block (4 waves) per batch element.  Output written fragment-tiled.
// ---------------------------------------------------------------------------
__global__ __launch_bounds__(128) void frontend_kernel(
    const float* __restrict__ prev_z, const float* __restrict__ u,
    const float* __restrict__ dt,
    const float* __restrict__ w_in, const float* __restrict__ b_in,
    const unsigned short* __restrict__ w_qkv_t, const float* __restrict__ b_qkv,
    const unsigned short* __restrict__ w_o_t, const float* __restrict__ b_o,
    const float* __restrict__ ln_g, const float* __restrict__ ln_b,
    unsigned short* __restrict__ xln_out) {
  __shared__ float          s_in[32];
  __shared__ unsigned short xfrag[4 * 512];
  __shared__ float          qkv[32 * 192];
  __shared__ float          attn[NHEAD * FIN * FIN];
  __shared__ unsigned short ofrag[4 * 512];
  __shared__ float          oo[32 * 64];

  const int b = blockIdx.x;
  const int tid = threadIdx.x;
  const int lane = tid & 31;
  const int wave = tid >> 5;

  if (tid < 32) {
    float v = 0.f;
    if (tid < LDIM)              v = prev_z[(size_t)b * LDIM + tid];
    else if (tid < LDIM + CDIM)  v = u[(size_t)b * CDIM + (tid - LDIM)];
    else if (tid == LDIM + CDIM) v = dt[b];
    s_in[tid] = v;
  }
  __syncthreads();

  for (int idx = tid; idx < 32 * 64; idx += 128) {
    const int i = idx >> 6, d = idx & 63;
    const float v = (i < FIN) ? (s_in[i] * w_in[d] + b_in[d]) : 0.f;
    const int sub = (i >> 4) * 2 + (d >> 5);
    xfrag[sub * 512 + a_lane(i, d & 31) * 16 + a_elem(d & 31)] = f2bf(v);
  }
  __syncthreads();

  // qkv = x @ w_qkv + b_qkv : (32x64)@(64x192)
  for (int t = wave; t < 24; t += 4) {
    const int tm = t / 12, tn = t % 12;
    v8f acc = v8f_zero();
#pragma unroll
    for (int ks = 0; ks < 2; ++ks) {
      const v16bf a  = load_frag(xfrag, tm * 2 + ks, lane);
      const v16bf bb = load_frag_bt(w_qkv_t, 64, ks * 32, tn * 16, lane);
      acc = wmma_bf16(a, bb, acc);
    }
    const int n = tn * 16 + (lane & 15);
    const int hi = lane >> 4;
    const float bias = b_qkv[n];
#pragma unroll
    for (int r = 0; r < 8; ++r) qkv[(tm * 16 + hi * 8 + r) * 192 + n] = acc[r] + bias;
  }
  __syncthreads();

  // attention: one head per wave; lane = key index, shuffle softmax
  {
    const int h = wave;
    for (int qi = 0; qi < FIN; ++qi) {
      float sc = -3.0e38f;
      if (lane < FIN) {
        float s = 0.f;
        const float* qrow = &qkv[qi * 192 + h * HDIM];
        const float* krow = &qkv[lane * 192 + DDIM + h * HDIM];
#pragma unroll
        for (int d = 0; d < HDIM; ++d) s += qrow[d] * krow[d];
        sc = s * 0.25f;
      }
      float mx = sc;
#pragma unroll
      for (int off = 16; off; off >>= 1) mx = fmaxf(mx, __shfl_xor(mx, off));
      const float e = (lane < FIN) ? __expf(sc - mx) : 0.f;
      float sm = e;
#pragma unroll
      for (int off = 16; off; off >>= 1) sm += __shfl_xor(sm, off);
      if (lane < FIN) attn[(h * FIN + qi) * FIN + lane] = e / sm;
    }
  }
  __syncthreads();

  for (int idx = tid; idx < 32 * 64; idx += 128) {
    const int qi = idx >> 6, d64 = idx & 63;
    float acc = 0.f;
    if (qi < FIN) {
      const int h = d64 >> 4, d = d64 & 15;
      const float* pr = &attn[(h * FIN + qi) * FIN];
      for (int kk = 0; kk < FIN; ++kk)
        acc += pr[kk] * qkv[kk * 192 + 2 * DDIM + h * HDIM + d];
    }
    const int sub = (qi >> 4) * 2 + (d64 >> 5);
    ofrag[sub * 512 + a_lane(qi, d64 & 31) * 16 + a_elem(d64 & 31)] = f2bf(acc);
  }
  __syncthreads();

  // oo = o @ w_o + b_o : (32x64)@(64x64)
  for (int t = wave; t < 8; t += 4) {
    const int tm = t >> 2, tn = t & 3;
    v8f acc = v8f_zero();
#pragma unroll
    for (int ks = 0; ks < 2; ++ks) {
      const v16bf a  = load_frag(ofrag, tm * 2 + ks, lane);
      const v16bf bb = load_frag_bt(w_o_t, 64, ks * 32, tn * 16, lane);
      acc = wmma_bf16(a, bb, acc);
    }
    const int n = tn * 16 + (lane & 15);
    const int hi = lane >> 4;
    const float bias = b_o[n];
#pragma unroll
    for (int r = 0; r < 8; ++r) oo[(tm * 16 + hi * 8 + r) * 64 + n] = acc[r] + bias;
  }
  __syncthreads();

  // layernorm over D=64 rows 0..24 -> fragment-tiled global (K = FEAT)
  for (int r = wave; r < FIN; r += 4) {
    const float v0 = oo[r * 64 + lane * 2];
    const float v1 = oo[r * 64 + lane * 2 + 1];
    float s = v0 + v1, s2 = v0 * v0 + v1 * v1;
#pragma unroll
    for (int off = 16; off; off >>= 1) { s += __shfl_xor(s, off); s2 += __shfl_xor(s2, off); }
    const float mu = s * (1.f / 64.f);
    const float var = s2 * (1.f / 64.f) - mu * mu;
    const float rs = rsqrtf(var + 1e-5f);
    const int d0 = lane * 2, d1 = d0 + 1;
    const int k0 = r * 64 + d0, k1 = r * 64 + d1;
    xln_out[afrag_idx((size_t)b, k0, FEAT)] = f2bf((v0 - mu) * rs * ln_g[d0] + ln_b[d0]);
    xln_out[afrag_idx((size_t)b, k1, FEAT)] = f2bf((v1 - mu) * rs * ln_g[d1] + ln_b[d1]);
  }
}

// ---------------------------------------------------------------------------
// Async-staged WMMA GEMM.  A and Bt are fragment-tiled in global memory, so
// each K-step tile is one contiguous 8KB (A) / 4KB (B) block copied with
// global_load_async_to_lds_b128 into double-buffered LDS, overlapped with WMMA.
// Block tile 128x64, 8 waves x (32x32).  M%128==0, K%32==0; N masked on store.
// ---------------------------------------------------------------------------
__global__ __launch_bounds__(256) void gemm_bf16_async_kernel(
    const unsigned short* __restrict__ A, const unsigned short* __restrict__ Bt,
    const float* __restrict__ bias, float* __restrict__ C,
    int M, int N, int K) {
  __shared__ unsigned short As[2][8 * 512];
  __shared__ unsigned short Bs[2][4 * 512];
  const int tid = threadIdx.x, lane = tid & 31, wave = tid >> 5;
  const int nkb = K >> 5;
  const size_t atile = (size_t)blockIdx.x * nkb;  // tile index base (x4096 ushort)
  const size_t btile = (size_t)blockIdx.y * nkb;  // tile index base (x2048 ushort)
  const int asub0 = (wave >> 1) * 2, bsub0 = (wave & 1) * 2;

  auto stage = [&](int kidx, int bufi) {
    const unsigned short* ga = A + (atile + kidx) * 4096;
    async_copy_b128(ga + tid * 8, &As[bufi][tid * 8]);
    async_copy_b128(ga + 2048 + tid * 8, &As[bufi][2048 + tid * 8]);
    const unsigned short* gb = Bt + (btile + kidx) * 2048;
    async_copy_b128(gb + tid * 8, &Bs[bufi][tid * 8]);
  };

  v8f acc00 = v8f_zero(), acc01 = v8f_zero(), acc10 = v8f_zero(), acc11 = v8f_zero();

  stage(0, 0);
  int buf = 0;
  for (int kidx = 0; kidx < nkb; ++kidx) {
    wait_async0();        // my async copies into current buffer are done
    __syncthreads();      // everyone's are; previous compute also finished
    if (kidx + 1 < nkb) stage(kidx + 1, buf ^ 1);

    const v16bf a0 = load_frag(As[buf], asub0, lane);
    const v16bf a1 = load_frag(As[buf], asub0 + 1, lane);
    const v16bf b0 = load_frag(Bs[buf], bsub0, lane);
    const v16bf b1 = load_frag(Bs[buf], bsub0 + 1, lane);
    acc00 = wmma_bf16(a0, b0, acc00);
    acc01 = wmma_bf16(a0, b1, acc01);
    acc10 = wmma_bf16(a1, b0, acc10);
    acc11 = wmma_bf16(a1, b1, acc11);
    buf ^= 1;
  }

  const int nlo = lane & 15, hi = lane >> 4;
  const int wm = (wave >> 1) * 32, wn = (wave & 1) * 32;
  const int bm = blockIdx.x * 128, bn = blockIdx.y * 64;
#pragma unroll
  for (int tm = 0; tm < 2; ++tm) {
#pragma unroll
    for (int tn = 0; tn < 2; ++tn) {
      const v8f acc = (tm == 0) ? (tn == 0 ? acc00 : acc01)
                                : (tn == 0 ? acc10 : acc11);
      const int n = bn + wn + tn * 16 + nlo;
      if (n >= N) continue;
      const float bv = bias ? bias[n] : 0.f;
#pragma unroll
      for (int r = 0; r < 8; ++r) {
        const int m = bm + wm + tm * 16 + hi * 8 + r;
        if (m < M) C[(size_t)m * N + n] = acc[r] + bv;
      }
    }
  }
}

// ---------------------------------------------------------------------------
// LayerNorm over HID=512, one wave per row, bf16 fragment-tiled output (K=512)
// ---------------------------------------------------------------------------
__global__ __launch_bounds__(128) void ln512_kernel(
    const float* __restrict__ x, const float* __restrict__ g,
    const float* __restrict__ bb, unsigned short* __restrict__ out) {
  const int lane = threadIdx.x & 31, wave = threadIdx.x >> 5;
  const size_t row = (size_t)blockIdx.x * 4 + wave;
  const float* xr = x + row * HID;
  float v[16];
  float s = 0.f, s2 = 0.f;
#pragma unroll
  for (int i = 0; i < 16; ++i) {
    v[i] = xr[lane + i * 32];
    s += v[i]; s2 += v[i] * v[i];
  }
#pragma unroll
  for (int off = 16; off; off >>= 1) { s += __shfl_xor(s, off); s2 += __shfl_xor(s2, off); }
  const float mu = s * (1.f / HID);
  const float var = s2 * (1.f / HID) - mu * mu;
  const float rs = rsqrtf(var + 1e-5f);
#pragma unroll
  for (int i = 0; i < 16; ++i) {
    const int d = lane + i * 32;
    out[afrag_idx(row, d, HID)] = f2bf((v[i] - mu) * rs * g[d] + bb[d]);
  }
}

// x = gelu(t) + 0.1*x ; also emit bf16 fragment-tiled copy of new x (K=512)
__global__ __launch_bounds__(256) void gelu_res_kernel(
    const float* __restrict__ t, float* __restrict__ x,
    unsigned short* __restrict__ xb, size_t n) {
  const size_t idx = (size_t)blockIdx.x * 256 + threadIdx.x;
  if (idx >= n) return;
  const float h = t[idx];
  const float ge = 0.5f * h * (1.f + erff(h * 0.70710678118f));
  const float nv = ge + 0.1f * x[idx];
  x[idx] = nv;
  xb[afrag_idx(idx >> 9, (int)(idx & 511), HID)] = f2bf(nv);
}

// ---------------------------------------------------------------------------
// Final: z_pred passthrough + 16x16 symmetric min-eig (power iter on cI - M)
// ---------------------------------------------------------------------------
__global__ __launch_bounds__(128) void cov_kernel(
    const float* __restrict__ outv, const float* __restrict__ Qk,
    float* __restrict__ zpred, float* __restrict__ cov) {
  const int tid = threadIdx.x;
  const int lane = tid & 31, wave = tid >> 5;
  const int sub = lane >> 4, i = lane & 15;
  const int b = blockIdx.x * 8 + wave * 2 + sub;

  const float* ob = outv + (size_t)b * 32;
  zpred[(size_t)b * LDIM + i] = ob[i];

  const float ls = ob[LDIM + i];
  const float* q = Qk + (size_t)b * 256;
  float mrow[16];
  float diagv = 0.f, rad = 0.f;
#pragma unroll
  for (int j = 0; j < 16; ++j) {
    const float dterm = (i == j) ? ls : 0.f;
    const float pij = 0.9f * softplusf(dterm + q[i * 16 + j]) + 0.1f;
    const float pji = 0.9f * softplusf(dterm + q[j * 16 + i]) + 0.1f;
    const float m = 0.5f * (pij + pji);
    mrow[j] = m;
    if (i == j) diagv = m; else rad += fabsf(m);
  }
  float ub = diagv + rad;
#pragma unroll
  for (int off = 8; off; off >>= 1) ub = fmaxf(ub, __shfl_xor(ub, off, 16));
  const float c = ub + 1e-3f;

  float x = 1.f + 0.03f * (float)i;
  {
    float nn = x * x;
#pragma unroll
    for (int off = 8; off; off >>= 1) nn += __shfl_xor(nn, off, 16);
    x *= rsqrtf(nn);
  }
  float lam = 0.f;
  for (int it = 0; it < 32; ++it) {
    float acc = 0.f;
#pragma unroll
    for (int j = 0; j < 16; ++j) acc += mrow[j] * __shfl(x, j, 16);
    const float y = c * x - acc;
    float dot = x * y, nn = y * y;
#pragma unroll
    for (int off = 8; off; off >>= 1) {
      dot += __shfl_xor(dot, off, 16);
      nn  += __shfl_xor(nn, off, 16);
    }
    lam = dot;
    x = y * rsqrtf(nn + 1e-30f);
  }
  const float min_e = c - lam;
  const float shift = fminf(min_e, 0.f);
#pragma unroll
  for (int j = 0; j < 16; ++j)
    cov[(size_t)b * 256 + (size_t)i * 16 + j] = mrow[j] - ((i == j) ? shift * 2.0f : 0.f);
}

// ---------------------------------------------------------------------------
extern "C" void kernel_launch(void* const* d_in, const int* in_sizes, int n_in,
                              void* d_out, int out_size, void* d_ws, size_t ws_size,
                              hipStream_t stream) {
  (void)in_sizes; (void)n_in; (void)out_size; (void)ws_size;
  const float* prev_z = (const float*)d_in[0];
  const float* u      = (const float*)d_in[1];
  const float* dt     = (const float*)d_in[2];
  const float* Qk     = (const float*)d_in[3];
  const float* w_in   = (const float*)d_in[4];
  const float* b_in   = (const float*)d_in[5];
  const float* w_qkv  = (const float*)d_in[6];
  const float* b_qkv  = (const float*)d_in[7];
  const float* w_o    = (const float*)d_in[8];
  const float* b_o    = (const float*)d_in[9];
  const float* ln_g   = (const float*)d_in[10];
  const float* ln_b   = (const float*)d_in[11];
  const float* w_flat = (const float*)d_in[12];
  const float* b_flat = (const float*)d_in[13];
  const float* m0_g   = (const float*)d_in[14];
  const float* m0_b   = (const float*)d_in[15];
  const float* m0_w   = (const float*)d_in[16];
  const float* m0_bw  = (const float*)d_in[17];
  const float* m1_g   = (const float*)d_in[18];
  const float* m1_b   = (const float*)d_in[19];
  const float* m1_w   = (const float*)d_in[20];
  const float* m1_bw  = (const float*)d_in[21];
  const float* w_out  = (const float*)d_in[22];
  const float* b_out  = (const float*)d_in[23];

  char* ws = (char*)d_ws;
  size_t off = 0;
  auto alloc = [&](size_t bytes) -> void* {
    void* p = ws + off;
    off = (off + bytes + 255) & ~(size_t)255;
    return p;
  };
  unsigned short* w_qkv_t  = (unsigned short*)alloc((size_t)192 * 64 * 2);
  unsigned short* w_o_t    = (unsigned short*)alloc((size_t)64 * 64 * 2);
  unsigned short* w_flat_f = (unsigned short*)alloc((size_t)HID * FEAT * 2);
  unsigned short* w_m0_f   = (unsigned short*)alloc((size_t)HID * HID * 2);
  unsigned short* w_m1_f   = (unsigned short*)alloc((size_t)HID * HID * 2);
  unsigned short* w_out_f  = (unsigned short*)alloc((size_t)64 * HID * 2); // N padded to 64
  unsigned short* xln_bf   = (unsigned short*)alloc((size_t)BATCH * FEAT * 2);
  float*          xbuf     = (float*)alloc((size_t)BATCH * HID * 4);
  unsigned short* hbf      = (unsigned short*)alloc((size_t)BATCH * HID * 2);
  float*          tbuf     = (float*)alloc((size_t)BATCH * HID * 4);
  float*          obuf     = (float*)alloc((size_t)BATCH * 32 * 4);

  cvt_bf16_t_kernel<<<(64 * 192 + 255) / 256, 256, 0, stream>>>(w_qkv, w_qkv_t, 64, 192);
  cvt_bf16_t_kernel<<<(64 * 64 + 255) / 256, 256, 0, stream>>>(w_o, w_o_t, 64, 64);
  cvt_bf16_frag_kernel<<<(FEAT * HID + 255) / 256, 256, 0, stream>>>(w_flat, w_flat_f, FEAT, HID);
  cvt_bf16_frag_kernel<<<(HID * HID + 255) / 256, 256, 0, stream>>>(m0_w, w_m0_f, HID, HID);
  cvt_bf16_frag_kernel<<<(HID * HID + 255) / 256, 256, 0, stream>>>(m1_w, w_m1_f, HID, HID);
  cvt_bf16_frag_kernel<<<(HID * 32 + 255) / 256, 256, 0, stream>>>(w_out, w_out_f, HID, 32);

  frontend_kernel<<<BATCH, 128, 0, stream>>>(prev_z, u, dt, w_in, b_in,
                                             w_qkv_t, b_qkv, w_o_t, b_o,
                                             ln_g, ln_b, xln_bf);

  // flatten GEMM: (B x 1600) @ (1600 x 512)
  gemm_bf16_async_kernel<<<dim3(BATCH / 128, HID / 64), 256, 0, stream>>>(
      xln_bf, w_flat_f, b_flat, xbuf, BATCH, HID, FEAT);

  // MLP layer 0
  ln512_kernel<<<BATCH / 4, 128, 0, stream>>>(xbuf, m0_g, m0_b, hbf);
  gemm_bf16_async_kernel<<<dim3(BATCH / 128, HID / 64), 256, 0, stream>>>(
      hbf, w_m0_f, m0_bw, tbuf, BATCH, HID, HID);
  gelu_res_kernel<<<(BATCH * HID) / 256, 256, 0, stream>>>(tbuf, xbuf, hbf,
                                                           (size_t)BATCH * HID);
  // MLP layer 1
  ln512_kernel<<<BATCH / 4, 128, 0, stream>>>(xbuf, m1_g, m1_b, hbf);
  gemm_bf16_async_kernel<<<dim3(BATCH / 128, HID / 64), 256, 0, stream>>>(
      hbf, w_m1_f, m1_bw, tbuf, BATCH, HID, HID);
  gelu_res_kernel<<<(BATCH * HID) / 256, 256, 0, stream>>>(tbuf, xbuf, hbf,
                                                           (size_t)BATCH * HID);

  // head: (B x 512) @ (512 x 32)  (B padded to 64 cols, stores masked)
  gemm_bf16_async_kernel<<<dim3(BATCH / 128, 1), 256, 0, stream>>>(
      hbf, w_out_f, b_out, obuf, BATCH, 32, HID);

  float* zout = (float*)d_out;
  float* cov  = zout + (size_t)BATCH * LDIM;
  cov_kernel<<<BATCH / 8, 128, 0, stream>>>(obuf, Qk, zout, cov);
}
